// EGTransformer_39719857554068
// MI455X (gfx1250) — matrix operand order repaired and checked by reading, not compiled, not run on hardware
//
#include <hip/hip_runtime.h>
#include <hip/hip_bf16.h>

// ---------------------------------------------------------------------------
// Deformable attention (EGTransformer) for MI455X / gfx1250, wave32 + WMMA.
// B=4, H=64, W=128, C=512, NH=8, NK=8, dk=64.
// All GEMMs share K = 512; M is templated so strides fold to immediates.
// ---------------------------------------------------------------------------

typedef __attribute__((ext_vector_type(16))) __bf16 v16bf;
typedef __attribute__((ext_vector_type(8)))  float  v8f;
typedef __attribute__((ext_vector_type(4)))  float  f32x4;
typedef __attribute__((ext_vector_type(4)))  __bf16 bf16x4;

#define DIM_B 4
#define DIM_H 64
#define DIM_W 128
#define DIM_C 512
#define DIM_NH 8
#define DIM_NK 8
#define DIM_DK 64
#define NTOK (DIM_B * DIM_H * DIM_W)      // 32768

// GEMM tiling
#define TILE_K 32
#define BLK_M 128
constexpr int GK = 512;                   // shared reduction dim

// native packed fp32->bf16 (RNE) + single 8-byte LDS store
__device__ __forceinline__ void cvt_store4(__bf16* dst, f32x4 v) {
    bf16x4 h = __builtin_convertvector(v, bf16x4);   // v_cvt_pk_bf16_f32 pairs
    *reinterpret_cast<unsigned long long*>(dst) =
        __builtin_bit_cast(unsigned long long, h);
}

// Y[N,MM] = X[N,512] @ Wt[512,MM] + bias  (fp32 in, bf16 WMMA, fp32 out)
// Block tile: 128 x BN; 8 waves as 4 row-groups x 2 col-groups;
// wave tile: 32 x (BN/2)  ->  2 x (BN/32) wmma 16x16x32 accumulators.
// Register-staged pipeline: next K-tile fetched (b128) while WMMAs run.
template<int MM, int BN>
__global__ __launch_bounds__(256)
void gemm_bias_wmma_bf16(const float* __restrict__ X,
                         const float* __restrict__ Wt,
                         const float* __restrict__ bias,
                         float* __restrict__ Y)
{
    constexpr int TN   = BN / 32;    // 16-col tiles per wave (2 or 4)
    constexpr int NF4A = 4;          // float4 per thread staging A (128x32)
    constexpr int NF4B = BN / 32;    // float4 per thread staging B (32xBN)

    __shared__ alignas(16) __bf16 As[BLK_M][TILE_K + 4];  // 72B rows, 8B-aligned
    __shared__ alignas(16) __bf16 Bs[TILE_K][BN + 4];

    const int tid  = threadIdx.x;
    const int lane = tid & 31;
    const int wave = tid >> 5;      // 0..7
    const int wm   = wave & 3;      // 4 wave-rows of 32
    const int wn   = wave >> 2;     // 2 wave-cols of BN/2

    const int rowBase = blockIdx.y * BLK_M;
    const int colBase = blockIdx.x * BN;

    const int mrow = lane & 15;
    const int hi   = lane >> 4;

    v8f acc[2][TN] = {};
    f32x4 ar[NF4A], br[NF4B];

    // ---- prologue: fetch K-tile 0 into registers (global_load_b128) ----
    #pragma unroll
    for (int i = 0; i < NF4A; ++i) {
        int e = i * 256 + tid, r = e >> 3, c4 = (e & 7) * 4;
        ar[i] = *reinterpret_cast<const f32x4*>(X + (size_t)(rowBase + r) * GK + c4);
    }
    #pragma unroll
    for (int i = 0; i < NF4B; ++i) {
        int e = i * 256 + tid, r = e / (BN / 4), c4 = (e % (BN / 4)) * 4;
        br[i] = *reinterpret_cast<const f32x4*>(Wt + (size_t)r * MM + colBase + c4);
    }

    #pragma unroll 1
    for (int k0 = 0; k0 < GK; k0 += TILE_K) {
        // ---- native cvt + packed ds_store_b64 of staged registers ----
        #pragma unroll
        for (int i = 0; i < NF4A; ++i) {
            int e = i * 256 + tid, r = e >> 3, c4 = (e & 7) * 4;
            cvt_store4(&As[r][c4], ar[i]);
        }
        #pragma unroll
        for (int i = 0; i < NF4B; ++i) {
            int e = i * 256 + tid, r = e / (BN / 4), c4 = (e % (BN / 4)) * 4;
            cvt_store4(&Bs[r][c4], br[i]);
        }
        __syncthreads();

        // ---- issue next K-tile global loads (latency hidden by wmma) ----
        if (k0 + TILE_K < GK) {
            int k1 = k0 + TILE_K;
            #pragma unroll
            for (int i = 0; i < NF4A; ++i) {
                int e = i * 256 + tid, r = e >> 3, c4 = (e & 7) * 4;
                ar[i] = *reinterpret_cast<const f32x4*>(
                            X + (size_t)(rowBase + r) * GK + k1 + c4);
            }
            #pragma unroll
            for (int i = 0; i < NF4B; ++i) {
                int e = i * 256 + tid, r = e / (BN / 4), c4 = (e % (BN / 4)) * 4;
                br[i] = *reinterpret_cast<const f32x4*>(
                            Wt + (size_t)(k1 + r) * MM + colBase + c4);
            }
        }

        // ---- fragments per ISA 7.12.2 layouts ----
        v16bf afrag[2], bfrag[TN];
        #pragma unroll
        for (int tm = 0; tm < 2; ++tm) {
            int r = wm * 32 + tm * 16 + mrow;          // A: lane holds row M
            #pragma unroll
            for (int v = 0; v < 8; ++v) {
                int kb = ((v >> 2) << 4) + (hi << 3) + ((v & 3) << 1);
                afrag[tm][2 * v]     = As[r][kb];
                afrag[tm][2 * v + 1] = As[r][kb + 1];
            }
        }
        #pragma unroll
        for (int tn = 0; tn < TN; ++tn) {
            int cb = wn * (16 * TN) + tn * 16;         // B: lane holds col N
            #pragma unroll
            for (int v = 0; v < 8; ++v) {
                int kb = ((v >> 2) << 4) + (hi << 3) + ((v & 3) << 1);
                bfrag[tn][2 * v]     = Bs[kb][cb + mrow];
                bfrag[tn][2 * v + 1] = Bs[kb + 1][cb + mrow];
            }
        }

        // ---- 2 x TN x v_wmma_f32_16x16x32_bf16 ----
        #pragma unroll
        for (int tm = 0; tm < 2; ++tm)
            #pragma unroll
            for (int tn = 0; tn < TN; ++tn)
                acc[tm][tn] = __builtin_amdgcn_wmma_f32_16x16x32_bf16(
                    false, afrag[tm], false, bfrag[tn],
                    (short)0, acc[tm][tn], false, false);

        __syncthreads();
    }

    // ---- epilogue: C/D layout M = v + 8*hi, N = lane%16; fuse bias ----
    #pragma unroll
    for (int tm = 0; tm < 2; ++tm)
        #pragma unroll
        for (int tn = 0; tn < TN; ++tn) {
            int col = colBase + wn * (16 * TN) + tn * 16 + mrow;
            float bv = bias[col];
            float* yp = Y + (size_t)(rowBase + wm * 32 + tm * 16 + 8 * hi) * MM + col;
            #pragma unroll
            for (int v = 0; v < 8; ++v)
                yp[(size_t)v * MM] = acc[tm][tn][v] + bv;
        }
}

// In-place softmax over the last NK=8 of [NTOK, NH, NK] (contiguous groups of 8)
__global__ __launch_bounds__(256)
void softmax_nk(float* __restrict__ A, int nGroups)
{
    int i = blockIdx.x * 256 + threadIdx.x;
    if (i >= nGroups) return;
    float* p = A + (size_t)i * DIM_NK;
    float m = p[0];
    #pragma unroll
    for (int k = 1; k < DIM_NK; ++k) m = fmaxf(m, p[k]);
    float e[DIM_NK], s = 0.f;
    #pragma unroll
    for (int k = 0; k < DIM_NK; ++k) { e[k] = __expf(p[k] - m); s += e[k]; }
    float inv = 1.f / s;
    #pragma unroll
    for (int k = 0; k < DIM_NK; ++k) p[k] = e[k] * inv;
}

__device__ __forceinline__ float tap(const float* __restrict__ kfb, int x, int y)
{
    if ((unsigned)x < (unsigned)DIM_W && (unsigned)y < (unsigned)DIM_H)
        return kfb[((size_t)y * DIM_W + x) * DIM_C];
    return 0.f;
}

// feat[b,h,w, nh*dk+ch] = sum_k A * bilinear(kf[b, ., ., nh*dk+ch], ref+off)
// One thread per (b,h,w,nh,ch); 64 channel-threads give coalesced gathers.
__global__ __launch_bounds__(256)
void deform_sample(const float* __restrict__ kf,
                   const float* __restrict__ off,
                   const float* __restrict__ A,
                   const float* __restrict__ ref,
                   float* __restrict__ feat)
{
    int idx = blockIdx.x * 256 + threadIdx.x;   // 24-bit index space
    int ch = idx & 63;
    int nh = (idx >> 6) & 7;
    int w  = (idx >> 9) & 127;
    int h  = (idx >> 16) & 63;
    int b  = idx >> 22;

    // reference's torch-.repeat quirk: ref batch = (b*NH + nh) % B
    int rb = (b * DIM_NH + nh) & (DIM_B - 1);

    const float* rp  = ref + (((size_t)rb * DIM_H + h) * DIM_W + w) * 2;
    const float refy = rp[0], refx = rp[1];

    size_t pix = ((size_t)b * DIM_H + h) * DIM_W + w;
    const float* op  = off + pix * (2 * DIM_NH * DIM_NK) + nh * (2 * DIM_NK);
    const float* ap  = A   + pix * (DIM_NH * DIM_NK)     + nh * DIM_NK;
    const float* kfb = kf  + (size_t)b * DIM_H * DIM_W * DIM_C + nh * DIM_DK + ch;

    // compose [-1,1] normalize + grid_sample(align_corners=False) un-normalize:
    // img = p * D/(D-1) - 0.5
    const float sy = (float)DIM_H / (float)(DIM_H - 1);
    const float sx = (float)DIM_W / (float)(DIM_W - 1);

    float acc = 0.f;
    #pragma unroll
    for (int k = 0; k < DIM_NK; ++k) {
        float py = refy + op[2 * k + 0];
        float px = refx + op[2 * k + 1];
        float yi = py * sy - 0.5f;
        float xi = px * sx - 0.5f;
        float y0f = floorf(yi), x0f = floorf(xi);
        float wy = yi - y0f,    wx = xi - x0f;
        int y0 = (int)y0f, x0 = (int)x0f;
        float v00 = tap(kfb, x0,     y0);
        float v10 = tap(kfb, x0 + 1, y0);
        float v01 = tap(kfb, x0,     y0 + 1);
        float v11 = tap(kfb, x0 + 1, y0 + 1);
        float s = v00 * (1.f - wx) * (1.f - wy) + v10 * wx * (1.f - wy)
                + v01 * (1.f - wx) * wy         + v11 * wx * wy;
        acc = fmaf(ap[k], s, acc);
    }
    feat[pix * DIM_C + nh * DIM_DK + ch] = acc;
}

// ---------------------------------------------------------------------------

extern "C" void kernel_launch(void* const* d_in, const int* in_sizes, int n_in,
                              void* d_out, int out_size, void* d_ws, size_t ws_size,
                              hipStream_t stream)
{
    (void)in_sizes; (void)n_in; (void)out_size; (void)ws_size;

    const float* query = (const float*)d_in[0];
    const float* keys0 = (const float*)d_in[1];
    const float* refpt = (const float*)d_in[2];
    const float* Wq    = (const float*)d_in[3];
    const float* bq    = (const float*)d_in[4];
    const float* Wk    = (const float*)d_in[5];
    const float* bk    = (const float*)d_in[6];
    const float* Woff  = (const float*)d_in[7];
    const float* boff  = (const float*)d_in[8];
    const float* WA    = (const float*)d_in[9];
    const float* bA    = (const float*)d_in[10];
    const float* Wm    = (const float*)d_in[11];
    const float* bm    = (const float*)d_in[12];

    // workspace partition (bytes); feat aliases q (q dead after off/A GEMMs)
    char* ws = (char*)d_ws;
    float* qbuf   = (float*)(ws);                                  // 64 MB
    float* featb  = qbuf;                                          // alias
    float* kfbuf  = (float*)(ws + (size_t)NTOK * DIM_C * 4);       // 64 MB
    float* offbuf = (float*)(ws + (size_t)NTOK * DIM_C * 8);       // 16 MB
    float* abuf   = (float*)(ws + (size_t)NTOK * DIM_C * 8
                                + (size_t)NTOK * 2 * DIM_NH * DIM_NK * 4); // 8 MB

    dim3 blk(256);
    const int NB = NTOK / BLK_M;    // 256 row blocks

    // 1. q = query @ Wq + bq          [32768, 512]
    gemm_bias_wmma_bf16<512, 128><<<dim3(4, NB), blk, 0, stream>>>(query, Wq, bq, qbuf);

    // 2. off = q @ Woff + boff        [32768, 128]
    gemm_bias_wmma_bf16<128, 128><<<dim3(1, NB), blk, 0, stream>>>(qbuf, Woff, boff, offbuf);

    // 3. A-logits = q @ WA + bA       [32768, 64]
    gemm_bias_wmma_bf16<64, 64><<<dim3(1, NB), blk, 0, stream>>>(qbuf, WA, bA, abuf);

    // 4. softmax over NK (in place; abuf fully rewritten each call above)
    {
        int groups = NTOK * DIM_NH;
        softmax_nk<<<dim3((groups + 255) / 256), blk, 0, stream>>>(abuf, groups);
    }

    // 5. kf = keys0 @ Wk + bk         [32768, 512]
    gemm_bias_wmma_bf16<512, 128><<<dim3(4, NB), blk, 0, stream>>>(keys0, Wk, bk, kfbuf);

    // 6. deformable bilinear sampling + weighted sum -> feat (aliases qbuf)
    {
        int total = DIM_B * DIM_H * DIM_W * DIM_NH * DIM_DK;   // 16.7M threads
        deform_sample<<<dim3(total / 256), blk, 0, stream>>>(
            kfbuf, offbuf, abuf, refpt, featb);
    }

    // 7. out = feat @ Wm + bm         [32768, 512]
    gemm_bias_wmma_bf16<512, 128><<<dim3(4, NB), blk, 0, stream>>>(featb, Wm, bm, (float*)d_out);
}
